// FieldDistillLossContrastive_28896539967631
// MI455X (gfx1250) — matrix operand order repaired and checked by reading, not compiled, not run on hardware
//
#include <hip/hip_runtime.h>

#define B_    8
#define N_    2048
#define M_    32
#define D_    768
#define DPF_  448
#define TOPK_ 10
#define BM_   (B_*M_)        // 256
#define MARGIN_ 0.3f
#define BIG_  1e18f
#define EPS_  1e-12f

typedef __attribute__((ext_vector_type(16))) __bf16 v16bf;
typedef __attribute__((ext_vector_type(8)))  float  v8f;

// accum layout (floats): [0..7] dist_sum/batch, [8..15] dist_cnt/batch,
// [16] loss_sum, [17] final_cnt, [18] texts_avg, [19] pts_avg
#define AC_DSUM 0
#define AC_DCNT 8
#define AC_LOSS 16
#define AC_FCNT 17
#define AC_TA   18
#define AC_PA   19

static __device__ __forceinline__ unsigned short f2bf(float f) {
  unsigned int u = __float_as_uint(f);
  unsigned int r = (u + 0x7fffu + ((u >> 16) & 1u)) >> 16;   // RNE
  return (unsigned short)r;
}
static __device__ __forceinline__ float bf2f(unsigned short h) {
  return __uint_as_float(((unsigned int)h) << 16);
}
// Build a v16bf fragment from two 16-byte contiguous chunks.
static __device__ __forceinline__ v16bf ld_frag(const unsigned short* p0,
                                                const unsigned short* p1) {
  union { uint4 q[2]; v16bf v; } u;
  u.q[0] = *(const uint4*)p0;
  u.q[1] = *(const uint4*)p1;
  return u.v;
}

// ---------------------------------------------------------------- init
__global__ void k_init(float* accum) {
  if (threadIdx.x < 32) accum[threadIdx.x] = 0.0f;
}

// ------------------------------------------------- prep: labels, norms, bf16 split
__global__ __launch_bounds__(256) void k_prep(
    const float* __restrict__ net, const float* __restrict__ pf,
    const float* __restrict__ mask, const float* __restrict__ coor,
    unsigned short* __restrict__ pf_hi, unsigned short* __restrict__ pf_lo,
    float* __restrict__ inv_net, int* __restrict__ label, float* __restrict__ sq) {
  int g = blockIdx.x * 256 + threadIdx.x;          // 0..B*N-1
  int b = g / N_, n = g % N_;
  int lab = -1;
  for (int m = M_ - 1; m >= 0; --m) {              // last mask containing point wins
    if (mask[((size_t)(b * M_ + m)) * N_ + n] > 0.0f) { lab = m; break; }
  }
  label[g] = lab;
  float x = coor[g * 3 + 0], y = coor[g * 3 + 1], z = coor[g * 3 + 2];
  sq[g] = x * x + y * y + z * z;

  const float* nr = net + (size_t)g * D_;
  float s = 0.f;
  for (int k = 0; k < D_; ++k) { float v = nr[k]; s += v * v; }
  inv_net[g] = 1.0f / fmaxf(sqrtf(s), EPS_);

  const float* pr = pf + (size_t)g * DPF_;
  s = 0.f;
  for (int k = 0; k < DPF_; ++k) { float v = pr[k]; s += v * v; }
  float ip = 1.0f / fmaxf(sqrtf(s), EPS_);
  unsigned short* ph = pf_hi + (size_t)g * DPF_;
  unsigned short* pl = pf_lo + (size_t)g * DPF_;
  for (int k = 0; k < DPF_; ++k) {
    float v = pr[k] * ip;
    unsigned short h = f2bf(v);
    ph[k] = h;
    pl[k] = f2bf(v - bf2f(h));
  }
}

// ---------------------------------------------------------------- npts
__global__ void k_npts(const float* __restrict__ mask, float* __restrict__ npts) {
  int bm = threadIdx.x;                            // 256 threads
  float s = 0.f;
  const float* mrow = mask + (size_t)bm * N_;
  for (int n = 0; n < N_; ++n) s += mrow[n];
  npts[bm] = s;
}

// ---------------------------------------------------- cast mask_embs to bf16 hi/lo
__global__ __launch_bounds__(256) void k_cast_me(
    const float* __restrict__ me, unsigned short* __restrict__ me_hi,
    unsigned short* __restrict__ me_lo) {
  size_t g = (size_t)blockIdx.x * 256 + threadIdx.x;   // BM*D = 196608
  float v = me[g];
  unsigned short h = f2bf(v);
  me_hi[g] = h;
  me_lo[g] = f2bf(v - bf2f(h));
}

// ------------------------------------------- avg_feats = (mask @ net)/npts, bf16 split
__global__ __launch_bounds__(256) void k_avgfeats(
    const float* __restrict__ net, const float* __restrict__ mask,
    const float* __restrict__ npts, unsigned short* __restrict__ af_hi,
    unsigned short* __restrict__ af_lo) {
  __shared__ float mp[M_][128];
  int b = blockIdx.x / 3, chunk = blockIdx.x % 3;
  int dd = chunk * 256 + threadIdx.x;
  float acc[M_];
#pragma unroll
  for (int m = 0; m < M_; ++m) acc[m] = 0.f;
  for (int p = 0; p < N_ / 128; ++p) {
    int nb = p * 128;
    for (int idx = threadIdx.x; idx < M_ * 128; idx += 256) {
      int m = idx >> 7, nn = idx & 127;
      mp[m][nn] = mask[((size_t)(b * M_ + m)) * N_ + nb + nn];
    }
    __syncthreads();
    for (int nn = 0; nn < 128; ++nn) {
      float v = net[((size_t)(b * N_ + nb + nn)) * D_ + dd];
#pragma unroll
      for (int m = 0; m < M_; ++m) acc[m] += mp[m][nn] * v;
    }
    __syncthreads();
  }
#pragma unroll
  for (int m = 0; m < M_; ++m) {
    float a = acc[m] / (npts[b * M_ + m] + EPS_);
    size_t g = (size_t)(b * M_ + m) * D_ + dd;
    unsigned short h = f2bf(a);
    af_hi[g] = h;
    af_lo[g] = f2bf(a - bf2f(h));
  }
}

// -------------------------------- logits = (me @ afT)*exp(ls) via split-bf16 WMMA
__global__ __launch_bounds__(32) void k_logits(
    const unsigned short* __restrict__ me_hi, const unsigned short* __restrict__ me_lo,
    const unsigned short* __restrict__ af_hi, const unsigned short* __restrict__ af_lo,
    const float* __restrict__ lsc, float* __restrict__ logits) {
  int lane = threadIdx.x;
  int p0 = blockIdx.y * 16, q0 = blockIdx.x * 16;
  int lh = lane >> 4, lr = lane & 15;
  const unsigned short* arh = me_hi + (size_t)(p0 + lr) * D_;
  const unsigned short* arl = me_lo + (size_t)(p0 + lr) * D_;
  const unsigned short* brh = af_hi + (size_t)(q0 + lr) * D_;
  const unsigned short* brl = af_lo + (size_t)(q0 + lr) * D_;
  v8f acc = {0, 0, 0, 0, 0, 0, 0, 0};
  for (int ks = 0; ks < D_; ks += 32) {
    int kb0 = ks + lh * 8, kb1 = ks + 16 + lh * 8, kc = ks + lh * 16;
    v16bf ah = ld_frag(arh + kb0, arh + kb1);
    v16bf al = ld_frag(arl + kb0, arl + kb1);
    v16bf bh = ld_frag(brh + kb0, brh + kb1);
    v16bf bl = ld_frag(brl + kb0, brl + kb1);
    acc = __builtin_amdgcn_wmma_f32_16x16x32_bf16(false, ah, false, bh, (short)0, acc, false, false);
    acc = __builtin_amdgcn_wmma_f32_16x16x32_bf16(false, ah, false, bl, (short)0, acc, false, false);
    acc = __builtin_amdgcn_wmma_f32_16x16x32_bf16(false, al, false, bh, (short)0, acc, false, false);
  }
  float es = expf(lsc[0]);
#pragma unroll
  for (int r = 0; r < 8; ++r) {
    int p = p0 + r + lh * 8;
    int q = q0 + lr;
    logits[p * BM_ + q] = acc[r] * es;
  }
}

// ------------------------------- row/col log-softmax diag losses + nonzero means
__global__ __launch_bounds__(256) void k_clip(
    const float* __restrict__ logits, const float* __restrict__ npts,
    float* __restrict__ accum) {
  __shared__ float s0[256], s1[256], s2[256], s3[256];
  int r = threadIdx.x;
  float mx = -3.4e38f;
  for (int c = 0; c < BM_; ++c) mx = fmaxf(mx, logits[r * BM_ + c]);
  float sm = 0.f;
  for (int c = 0; c < BM_; ++c) sm += expf(logits[r * BM_ + c] - mx);
  float lse_r = mx + logf(sm);
  mx = -3.4e38f;
  for (int c = 0; c < BM_; ++c) mx = fmaxf(mx, logits[c * BM_ + r]);
  sm = 0.f;
  for (int c = 0; c < BM_; ++c) sm += expf(logits[c * BM_ + r] - mx);
  float lse_c = mx + logf(sm);
  float diag = logits[r * (BM_ + 1)];
  bool valid = npts[r] > 0.f;
  float tl = valid ? (lse_r - diag) : 0.f;
  float pl = valid ? (lse_c - diag) : 0.f;
  s0[r] = tl; s1[r] = (tl > 0.f) ? 1.f : 0.f;
  s2[r] = pl; s3[r] = (pl > 0.f) ? 1.f : 0.f;
  __syncthreads();
  for (int off = 128; off > 0; off >>= 1) {
    if (r < off) { s0[r] += s0[r + off]; s1[r] += s1[r + off];
                   s2[r] += s2[r + off]; s3[r] += s3[r + off]; }
    __syncthreads();
  }
  if (r == 0) {
    accum[AC_TA] = (s0[0] > 0.f) ? s0[0] / fmaxf(s1[0], 1.f) : 0.f;
    accum[AC_PA] = (s2[0] > 0.f) ? s2[0] / fmaxf(s3[0], 1.f) : 0.f;
  }
}

// --------- fused pf-gram (split-bf16 WMMA) + per-row dual top-k + distance sums
__global__ __launch_bounds__(128) void k_partscan(
    const unsigned short* __restrict__ pf_hi, const unsigned short* __restrict__ pf_lo,
    const int* __restrict__ label, const float* __restrict__ sq,
    const float* __restrict__ coor, int* __restrict__ tidx_o,
    int* __restrict__ nidx_o, float* __restrict__ ntd_o,
    int* __restrict__ pv_o, float* __restrict__ accum) {
  __shared__ unsigned short aHi[16 * DPF_];
  __shared__ unsigned short aLo[16 * DPF_];
  __shared__ float panel[16][65];                  // pad to avoid bank conflicts

  const int tid = threadIdx.x;
  const int wave = tid >> 5, lane = tid & 31;
  const int lh = lane >> 4, lr = lane & 15;
  const int b = blockIdx.y;
  const int i0 = blockIdx.x * 16;
  const int base = b * N_;

  // stage the 16 A rows (hi+lo) into LDS: rows contiguous in global memory
  {
    const uint4* gh = (const uint4*)(pf_hi + (size_t)(base + i0) * DPF_);
    const uint4* gl = (const uint4*)(pf_lo + (size_t)(base + i0) * DPF_);
    uint4* sh = (uint4*)aHi;
    uint4* sl = (uint4*)aLo;
    for (int c = tid; c < 16 * DPF_ * 2 / 16; c += 128) { sh[c] = gh[c]; sl[c] = gl[c]; }
  }
  __syncthreads();

  // per-row scan state (meaningful on tid<16)
  float tv[TOPK_]; int ti_[TOPK_];
  float nv[TOPK_]; int ni_[TOPK_];
  float tvmin = -1e30f; int tvmin_ix = 0;
  float nvmax = BIG_;   int nvmax_ix = 0;
  float row_dsum = 0.f, row_dcnt = 0.f;
  int li = -1; float sqi = 0.f, cix = 0.f, ciy = 0.f, ciz = 0.f;
#pragma unroll
  for (int q = 0; q < TOPK_; ++q) { tv[q] = -1e30f; ti_[q] = 0; nv[q] = BIG_; ni_[q] = q; }
  if (tid < 16) {
    int i = i0 + tid;
    li = label[base + i];
    sqi = sq[base + i];
    cix = coor[(size_t)(base + i) * 3 + 0];
    ciy = coor[(size_t)(base + i) * 3 + 1];
    ciz = coor[(size_t)(base + i) * 3 + 2];
  }

  const unsigned short* arh = aHi + lr * DPF_;
  const unsigned short* arl = aLo + lr * DPF_;

  for (int p = 0; p < N_ / 64; ++p) {
    const int jbase = p * 64;
    const int c0 = jbase + wave * 16;
    const unsigned short* brh = pf_hi + (size_t)(base + c0 + lr) * DPF_;
    const unsigned short* brl = pf_lo + (size_t)(base + c0 + lr) * DPF_;
    if (p + 1 < N_ / 64)
      __builtin_prefetch(pf_hi + (size_t)(base + jbase + 64 + wave * 16 + lr) * DPF_, 0, 0);

    v8f acc = {0, 0, 0, 0, 0, 0, 0, 0};
    for (int ks = 0; ks < DPF_; ks += 32) {
      int kb0 = ks + lh * 8, kb1 = ks + 16 + lh * 8, kc = ks + lh * 16;
      v16bf ah = ld_frag(arh + kb0, arh + kb1);
      v16bf al = ld_frag(arl + kb0, arl + kb1);
      v16bf bh = ld_frag(brh + kb0, brh + kb1);
      v16bf bl = ld_frag(brl + kb0, brl + kb1);
      acc = __builtin_amdgcn_wmma_f32_16x16x32_bf16(false, ah, false, bh, (short)0, acc, false, false);
      acc = __builtin_amdgcn_wmma_f32_16x16x32_bf16(false, ah, false, bl, (short)0, acc, false, false);
      acc = __builtin_amdgcn_wmma_f32_16x16x32_bf16(false, al, false, bh, (short)0, acc, false, false);
    }
    __syncthreads();                               // previous panel fully scanned
#pragma unroll
    for (int r = 0; r < 8; ++r) {
      panel[r + lh * 8][wave * 16 + lr] = acc[r];  // C layout: M=r+8*lh, N=lr
    }
    __syncthreads();

    if (tid < 16) {
      const int i = i0 + tid;
      for (int jj = 0; jj < 64; ++jj) {
        const int j = jbase + jj;
        const float s = panel[tid][jj];
        const int lj = label[base + j];
        // same-label positive candidate (ref: pf_sim * same, 0 elsewhere)
        const bool same = (lj == li) && (lj != -1) && (j != i);
        const float cand = same ? s : 0.0f;
        if (cand > tvmin) {
#pragma unroll
          for (int q = 0; q < TOPK_; ++q)
            if (q == tvmin_ix) { tv[q] = cand; ti_[q] = j; }
          tvmin = tv[0]; tvmin_ix = 0;
#pragma unroll
          for (int q = 1; q < TOPK_; ++q)
            if (tv[q] < tvmin) { tvmin = tv[q]; tvmin_ix = q; }
        }
        // different-label distance candidate
        const bool diff = (lj != li) && (lj != -1);
        if (diff) {
          const float cjx = coor[(size_t)(base + j) * 3 + 0];
          const float cjy = coor[(size_t)(base + j) * 3 + 1];
          const float cjz = coor[(size_t)(base + j) * 3 + 2];
          const float dot = cix * cjx + ciy * cjy + ciz * cjz;
          const float d = sqrtf(fmaxf(sqi + sq[base + j] - 2.0f * dot, 0.0f));
          row_dsum += d;
          row_dcnt += 1.0f;
          if (d < nvmax) {
#pragma unroll
            for (int q = 0; q < TOPK_; ++q)
              if (q == nvmax_ix) { nv[q] = d; ni_[q] = j; }
            nvmax = nv[0]; nvmax_ix = 0;
#pragma unroll
            for (int q = 1; q < TOPK_; ++q)
              if (nv[q] > nvmax) { nvmax = nv[q]; nvmax_ix = q; }
          }
        }
      }
    }
  }

  if (tid < 16) {
    const int i = i0 + tid;
    const size_t g = (size_t)(base + i);
    int cnt3 = 0;
#pragma unroll
    for (int q = 0; q < TOPK_; ++q) cnt3 += (tv[q] > 0.5f) ? 1 : 0;
    const int pv = (cnt3 >= 3 && li != -1) ? 1 : 0;
#pragma unroll
    for (int q = 0; q < TOPK_; ++q) {
      tidx_o[g * TOPK_ + q] = ti_[q];
      nidx_o[g * TOPK_ + q] = ni_[q];
      ntd_o[g * TOPK_ + q]  = nv[q];
    }
    pv_o[g] = pv;
    if (pv) {
      atomicAdd(&accum[AC_DSUM + b], row_dsum);
      atomicAdd(&accum[AC_DCNT + b], row_dcnt);
    }
  }
}

// ---------------- per-row final: margin loss with lazy net_sim at gathered indices
__global__ __launch_bounds__(256) void k_partfinal(
    const float* __restrict__ net, const float* __restrict__ inv_net,
    const int* __restrict__ tidx, const int* __restrict__ nidx,
    const float* __restrict__ ntd, const int* __restrict__ pv,
    const float* __restrict__ accum, float* __restrict__ accw) {
  const int wave = threadIdx.x >> 5, lane = threadIdx.x & 31;
  const int g = blockIdx.x * 8 + wave;             // 0..B*N-1
  const int b = g / N_;
  if (!pv[g]) return;
  const float cs = accum[AC_DCNT + b];
  const float dmean = (cs > 0.f) ? accum[AC_DSUM + b] / fmaxf(cs, 1.f) : BIG_;
  int cnt = 0;
#pragma unroll
  for (int q = 0; q < TOPK_; ++q) cnt += (ntd[(size_t)g * TOPK_ + q] < dmean) ? 1 : 0;
  if (cnt < 3) return;

  const float* xi = net + (size_t)g * D_;
  float posacc = 0.f, negacc = 0.f;
#pragma unroll
  for (int q = 0; q < TOPK_; ++q) {
    const int jp = tidx[(size_t)g * TOPK_ + q];
    const float* xj = net + (size_t)(b * N_ + jp) * D_;
    float pp = 0.f;
    for (int d0 = lane; d0 < D_; d0 += 32) pp += xi[d0] * xj[d0];
    posacc += pp * inv_net[b * N_ + jp];
    const int jn = nidx[(size_t)g * TOPK_ + q];
    const float* xk = net + (size_t)(b * N_ + jn) * D_;
    float nn = 0.f;
    for (int d0 = lane; d0 < D_; d0 += 32) nn += xi[d0] * xk[d0];
    negacc += nn * inv_net[b * N_ + jn];
  }
#pragma unroll
  for (int off = 16; off > 0; off >>= 1) {
    posacc += __shfl_xor(posacc, off, 32);
    negacc += __shfl_xor(negacc, off, 32);
  }
  if (lane == 0) {
    const float invi = inv_net[g];
    const float pos_sim = posacc * invi * (1.0f / TOPK_);
    const float neg_sim = negacc * invi * (1.0f / TOPK_);
    const float loss = fmaxf(MARGIN_ - (pos_sim - neg_sim), 0.0f);
    atomicAdd(&accw[AC_LOSS], loss);
    atomicAdd(&accw[AC_FCNT], 1.0f);
  }
}

// ---------------------------------------------------------------- combine
__global__ void k_final(const float* __restrict__ accum, float* __restrict__ out) {
  const float ls = accum[AC_LOSS], fc = accum[AC_FCNT];
  const float part = (fc > 0.f) ? ls / fmaxf(fc, 1.f) : 0.f;
  out[0] = 0.5f * (accum[AC_TA] + accum[AC_PA]) + 1.0f * part;
}

// ---------------------------------------------------------------- launch
extern "C" void kernel_launch(void* const* d_in, const int* in_sizes, int n_in,
                              void* d_out, int out_size, void* d_ws, size_t ws_size,
                              hipStream_t stream) {
  (void)in_sizes; (void)n_in; (void)out_size; (void)ws_size;
  const float* net  = (const float*)d_in[0];   // [B*N, D]
  const float* me   = (const float*)d_in[1];   // [B*M, D]
  const float* mask = (const float*)d_in[2];   // [B, M, N]
  const float* pf   = (const float*)d_in[3];   // [B*N, DPF]
  const float* coor = (const float*)d_in[4];   // [B*N, 3]
  const float* lsc  = (const float*)d_in[5];   // scalar
  float* out = (float*)d_out;

  char* ws = (char*)d_ws;
  constexpr size_t SZ_PF  = (size_t)B_ * N_ * DPF_ * 2;
  constexpr size_t SZ_BN4 = (size_t)B_ * N_ * 4;
  constexpr size_t SZ_ME  = (size_t)BM_ * D_ * 2;
  constexpr size_t SZ_TK  = (size_t)B_ * N_ * TOPK_ * 4;
  size_t off = 0;
  unsigned short* pf_hi = (unsigned short*)(ws + off); off += SZ_PF;
  unsigned short* pf_lo = (unsigned short*)(ws + off); off += SZ_PF;
  float* inv_net = (float*)(ws + off); off += SZ_BN4;
  int*   label   = (int*)(ws + off);   off += SZ_BN4;
  float* sq      = (float*)(ws + off); off += SZ_BN4;
  unsigned short* me_hi = (unsigned short*)(ws + off); off += SZ_ME;
  unsigned short* me_lo = (unsigned short*)(ws + off); off += SZ_ME;
  unsigned short* af_hi = (unsigned short*)(ws + off); off += SZ_ME;
  unsigned short* af_lo = (unsigned short*)(ws + off); off += SZ_ME;
  float* npts   = (float*)(ws + off); off += (size_t)BM_ * 4;
  float* logits = (float*)(ws + off); off += (size_t)BM_ * BM_ * 4;
  int*   tidx   = (int*)(ws + off);   off += SZ_TK;
  int*   nidx   = (int*)(ws + off);   off += SZ_TK;
  float* ntd    = (float*)(ws + off); off += SZ_TK;
  int*   pv     = (int*)(ws + off);   off += SZ_BN4;
  float* accum  = (float*)(ws + off); off += 128;

  k_init<<<1, 32, 0, stream>>>(accum);
  k_prep<<<(B_ * N_) / 256, 256, 0, stream>>>(net, pf, mask, coor,
                                              pf_hi, pf_lo, inv_net, label, sq);
  k_npts<<<1, 256, 0, stream>>>(mask, npts);
  k_cast_me<<<(BM_ * D_) / 256, 256, 0, stream>>>(me, me_hi, me_lo);
  k_avgfeats<<<B_ * 3, 256, 0, stream>>>(net, mask, npts, af_hi, af_lo);
  k_logits<<<dim3(BM_ / 16, BM_ / 16), 32, 0, stream>>>(me_hi, me_lo, af_hi, af_lo,
                                                        lsc, logits);
  k_clip<<<1, 256, 0, stream>>>(logits, npts, accum);
  k_partscan<<<dim3(N_ / 16, B_), 128, 0, stream>>>(pf_hi, pf_lo, label, sq, coor,
                                                    tidx, nidx, ntd, pv, accum);
  k_partfinal<<<(B_ * N_) / 8, 256, 0, stream>>>(net, inv_net, tidx, nidx, ntd, pv,
                                                 accum, accum);
  k_final<<<1, 1, 0, stream>>>(accum, out);
}
